// StokenAttention_10780367913137
// MI455X (gfx1250) — matrix operand order repaired and checked by reading, not compiled, not run on hardware
//
#include <hip/hip_runtime.h>

// ---------------------------------------------------------------------------
// StokenAttention forward for MI455X (gfx1250, wave32, WMMA bf16 path)
// B-side operands stored pair-packed ([K/2][N] dwords) so fragment loads are
// dword loads matching the WMMA VGPR layout directly (no b16 packing movs).
// ---------------------------------------------------------------------------

typedef __bf16 bf16;
typedef bf16  v16bf __attribute__((ext_vector_type(16)));
typedef bf16  v8bf  __attribute__((ext_vector_type(8)));
typedef float v8f   __attribute__((ext_vector_type(8)));
typedef unsigned int uint32;

#define DEV __device__ __forceinline__

// geometry
constexpr int BN   = 4;
constexpr int CH   = 384;
constexpr int HW   = 224;
constexpr int GH   = 28;      // hh == ww
constexpr int NSP  = 784;     // 28*28 superpixels
constexpr int HEADS = 8;
constexpr int HD   = 48;
constexpr int C3   = 1152;    // 3*C
constexpr float SCALE_AFF = 0.05103103630798288f;   // 384^-0.5
constexpr float SCALE_ATT = 0.14433756729740643f;   // 48^-0.5

// ---------------- workspace layout (bytes) ----------------
constexpr size_t aln(size_t x) { return (x + 255) & ~(size_t)255; }

constexpr size_t SZ_PIXB  = (size_t)BN * NSP * CH * 64 * 2;   // bf16 pix^T (A layout)
constexpr size_t OFF_PIXB = 0;
constexpr size_t OFF_SF   = aln(OFF_PIXB + SZ_PIXB);
constexpr size_t SZ_SF    = (size_t)BN * NSP * CH * 4;        // f32 (b,n,c)
constexpr size_t OFF_AFFP = aln(OFF_SF + SZ_SF);
constexpr size_t SZ_AFFP  = (size_t)BN * NSP * 64 * 16 * 2;   // bf16 (p, kk-padded)
constexpr size_t OFF_AFFT = aln(OFF_AFFP + SZ_AFFP);
constexpr size_t SZ_AFFT  = (size_t)BN * NSP * 32 * 64 * 2;   // bf16 (kk padded to 32, p)
constexpr size_t OFF_AS9  = aln(OFF_AFFT + SZ_AFFT);
constexpr size_t SZ_AS9   = (size_t)BN * NSP * 9 * 4;
constexpr size_t OFF_ASUM = aln(OFF_AS9 + SZ_AS9);
constexpr size_t SZ_ASUM  = (size_t)BN * NSP * 4;
constexpr size_t OFF_SFA  = aln(OFF_ASUM + SZ_ASUM);
constexpr size_t SZ_SFA   = (size_t)BN * CH * NSP * 4;        // f32 fold accumulator
constexpr size_t OFF_SFABP= aln(OFF_SFA + SZ_SFA);
constexpr size_t SZ_SFABP = (size_t)BN * 192 * NSP * 4;       // dword pairs (b,c/2,pos)
constexpr size_t OFF_QW   = aln(OFF_SFABP + SZ_SFABP);
constexpr size_t SZ_QW    = (size_t)C3 * CH * 2;
constexpr size_t OFF_PW   = aln(OFF_QW + SZ_QW);
constexpr size_t SZ_PW    = (size_t)CH * CH * 2;
constexpr size_t OFF_QKVB = aln(OFF_PW + SZ_PW);
constexpr size_t SZ_QKVB  = (size_t)BN * C3 * NSP * 2;        // bf16 row-major
constexpr size_t OFF_QKVP = aln(OFF_QKVB + SZ_QKVB);
constexpr size_t SZ_QKVP  = (size_t)BN * 576 * NSP * 4;       // dword pairs (b,row/2,n)
constexpr size_t OFF_CSUM = aln(OFF_QKVP + SZ_QKVP);
constexpr size_t SZ_CSUM  = (size_t)BN * HEADS * NSP * 4;
constexpr size_t OFF_OUTHP= aln(OFF_CSUM + SZ_CSUM);
constexpr size_t SZ_OUTHP = (size_t)BN * 192 * NSP * 4;       // dword pairs (b,c/2,n)
constexpr size_t OFF_OUTP = aln(OFF_OUTHP + SZ_OUTHP);
constexpr size_t SZ_OUTP  = (size_t)BN * CH * NSP * 2;        // bf16 row-major
constexpr size_t OFF_PIXP = aln(OFF_OUTP + SZ_OUTP);
constexpr size_t SZ_PIXP  = (size_t)BN * NSP * 192 * 64 * 4;  // dword pairs (b,n,c/2,p)
// attention scratch aliased over PIXB (dead after k_sft):
constexpr size_t OFF_ATTNS = 0;                               // f32 32*784*784 = 78.7 MB
constexpr size_t OFF_ATTNE = (size_t)80 << 20;                // dwords 32*400*784 = 40.1 MB

// ---------------- WMMA fragment helpers (CDNA5 layouts) ----------------
DEV v8f vzero() { v8f z = {0.f,0.f,0.f,0.f,0.f,0.f,0.f,0.f}; return z; }

DEV uint32 pack2(float lo, float hi) {
  union { uint32 u; bf16 h[2]; } pk;
  pk.h[0] = (bf16)lo; pk.h[1] = (bf16)hi;
  return pk.u;
}

// A-matrix 16x32 bf16, generic strided variant: lane l -> m = l&15, half = l>>4.
// VGPR v, slot s: k = ((v>>2)<<4) + half*8 + ((v&3)<<1) + s
DEV v16bf frag_a(const bf16* p, long sm, long sk, int kmax, int lane) {
  int m = lane & 15, half = lane >> 4;
  v16bf r;
#pragma unroll
  for (int v = 0; v < 8; ++v) {
#pragma unroll
    for (int s = 0; s < 2; ++s) {
      int k = ((v >> 2) << 4) + half * 8 + ((v & 3) << 1) + s;
      r[2 * v + s] = (k < kmax) ? p[(long)m * sm + (long)k * sk] : (bf16)0.0f;
    }
  }
  return r;
}

// A-matrix, K-contiguous (sk==1): one dword per packed pair. kmax even; 4B aligned rows.
DEV v16bf frag_a1(const bf16* p, long sm, int kmax, int lane) {
  int m = lane & 15, half = lane >> 4;
  union { v16bf bv; uint32 u[8]; } r;
  const uint32* q = (const uint32*)(p + (long)m * sm);
#pragma unroll
  for (int v = 0; v < 8; ++v) {
    int kp = ((v >> 2) << 3) + half * 4 + (v & 3);
    r.u[v] = ((2 * kp) < kmax) ? q[kp] : 0u;
  }
  return r.bv;
}

// A-matrix from pair-packed layout: element (m,k) at dword q[(k/2)*skdw + m*sm].
DEV v16bf frag_a_pl(const uint32* q, long sm, long skdw, int kmax, int lane) {
  int m = lane & 15, half = lane >> 4;
  union { v16bf bv; uint32 u[8]; } r;
#pragma unroll
  for (int v = 0; v < 8; ++v) {
    int kp = ((v >> 2) << 3) + half * 4 + (v & 3);
    r.u[v] = ((2 * kp) < kmax) ? q[(long)kp * skdw + (long)m * sm] : 0u;
  }
  return r.bv;
}

// B-matrix 32x16 bf16 from scalar layout (small reused fragments only).
DEV v16bf frag_b(const bf16* p, long sk, long sn, int kmax, int lane) {
  int n = lane & 15, half = lane >> 4;
  v16bf r;
#pragma unroll
  for (int e = 0; e < 16; ++e) {
    int k = (e < 8) ? (half * 8 + e) : (16 + half * 8 + (e - 8));
    r[e] = (k < kmax) ? p[(long)k * sk + (long)n * sn] : (bf16)0.0f;
  }
  return r;
}

// B-matrix from pair-packed layout: VGPR v holds K = ((v>>2)<<4)+half*8+((v&3)<<1), +1 -> dword q[kp*sndw + n].
DEV v16bf frag_bp(const uint32* q, long sndw, int kmax, int lane) {
  int n = lane & 15, half = lane >> 4;
  union { v16bf bv; uint32 u[8]; } r;
#pragma unroll
  for (int v = 0; v < 8; ++v) {
    int kp = ((v >> 2) << 3) + half * 4 + (v & 3);
    r.u[v] = ((2 * kp) < kmax) ? q[(long)kp * sndw + n] : 0u;
  }
  return r.bv;
}

DEV v8f wmma_bf16(v16bf a, v16bf b, v8f c) {
  return __builtin_amdgcn_wmma_f32_16x16x32_bf16(false, a, false, b, (short)0, c, false, false);
}
// C/D layout: lane l -> n = l&15, hi = l>>4; VGPR v -> m = v + 8*hi

// ---------------- Tensor Data Mover staging (guarded, both toolchains) -----
#if __has_builtin(__builtin_amdgcn_tensor_load_to_lds) && \
    __has_builtin(__builtin_amdgcn_s_wait_tensorcnt)
#define HAVE_TDM 1
typedef unsigned int u32x4 __attribute__((ext_vector_type(4)));
typedef int i32x4 __attribute__((ext_vector_type(4)));
typedef int i32x8 __attribute__((ext_vector_type(8)));
// DMA one contiguous 2048-element (4 KB) bf16 tile from global to LDS offset lds_off.
DEV void tdm_load_tile_4k(uint32 lds_off, const void* gsrc) {
  unsigned long long ga = (unsigned long long)(uintptr_t)gsrc;
  u32x4 g0;
  g0[0] = 1u;                                               // count=1, user mode
  g0[1] = lds_off;                                          // lds_addr (bytes)
  g0[2] = (uint32)(ga & 0xffffffffu);                       // global_addr[31:0]
  g0[3] = (uint32)((ga >> 32) & 0x01ffffffu) | (2u << 30);  // global_addr[56:32] | type=2
  i32x8 g1;
  g1[0] = 0x10000;              // data_size = 1 (2 bytes/elem)
  g1[1] = (int)(2048u << 16);   // tensor_dim0 = 2048 (bits 63:48)
  g1[2] = (int)(1u << 16);      // tensor_dim1 = 1     (bits 111:80)
  g1[3] = (int)(2048u << 16);   // tile_dim0 = 2048    (bits 127:112)
  g1[4] = 0;                    // tile_dim1/tile_dim2 unused
  g1[5] = 2048;                 // tensor_dim0_stride = 2048
  g1[6] = 0;
  g1[7] = 0;
  i32x4 gz = {0, 0, 0, 0};
#if __clang_major__ >= 23
  i32x8 gz8 = {0, 0, 0, 0, 0, 0, 0, 0};
  __builtin_amdgcn_tensor_load_to_lds(g0, g1, gz, gz, gz8, 0);
#else
  __builtin_amdgcn_tensor_load_to_lds(g0, g1, gz, gz, 0);
#endif
  __builtin_amdgcn_s_wait_tensorcnt(0);
}
#endif

// ---------------- kernels ----------------

__global__ void k_zero(float* __restrict__ p, long nElem) {
  long i = (long)blockIdx.x * blockDim.x + threadIdx.x;
  if (i < nElem) p[i] = 0.f;
}

// pool + transpose: xs(B,C,H,W) -> sf(B,n,C), pixb(B,n,C,64) bf16, pixP(B,n,C/2,64) dword pairs
__global__ void k_pool(const float* __restrict__ xs, float* __restrict__ sf,
                       bf16* __restrict__ pixb, uint32* __restrict__ pixP) {
  long idx = (long)blockIdx.x * blockDim.x + threadIdx.x;
  if (idx < (long)BN * 192 * NSP) {
    int nidx = (int)(idx % NSP);
    int kp   = (int)((idx / NSP) % 192);
    int b    = (int)(idx / ((long)NSP * 192));
    int c0 = 2 * kp;
    int y = nidx / GH, x = nidx % GH;
    const float* src0 = xs + (((long)(b * CH + c0) * HW + y * 8) * HW + x * 8);
    const float* src1 = src0 + (long)HW * HW;
    bf16* dstA = pixb + (((long)(b * NSP + nidx) * CH + c0) << 6);
    uint32* dstP = pixP + (((long)(b * NSP + nidx) * 192 + kp) << 6);
    float s0 = 0.f, s1 = 0.f;
#pragma unroll
    for (int r = 0; r < 8; ++r) {
      const float4* p0 = reinterpret_cast<const float4*>(src0 + (long)r * HW);
      const float4* p1 = reinterpret_cast<const float4*>(src1 + (long)r * HW);
      float4 a0 = p0[0], b0 = p0[1], a1 = p1[0], b1 = p1[1];
      s0 += a0.x + a0.y + a0.z + a0.w + b0.x + b0.y + b0.z + b0.w;
      s1 += a1.x + a1.y + a1.z + a1.w + b1.x + b1.y + b1.z + b1.w;
      float f0[8] = {a0.x, a0.y, a0.z, a0.w, b0.x, b0.y, b0.z, b0.w};
      float f1[8] = {a1.x, a1.y, a1.z, a1.w, b1.x, b1.y, b1.z, b1.w};
      v8bf o0, o1;
#pragma unroll
      for (int j = 0; j < 8; ++j) {
        o0[j] = (bf16)f0[j];
        o1[j] = (bf16)f1[j];
        dstP[r * 8 + j] = pack2(f0[j], f1[j]);
      }
      *reinterpret_cast<v8bf*>(dstA + r * 8) = o0;
      *reinterpret_cast<v8bf*>(dstA + 64 + r * 8) = o1;
    }
    sf[(long)(b * NSP + nidx) * CH + c0] = s0 * (1.f / 64.f);
    sf[(long)(b * NSP + nidx) * CH + c0 + 1] = s1 * (1.f / 64.f);
  }
#if __has_builtin(__builtin_amdgcn_cluster_id_x) && \
    __has_builtin(__builtin_amdgcn_s_cluster_barrier)
  // Cluster-aware sync point: NOP when not dispatched in clusters (ISA 3.2).
  if (__builtin_amdgcn_cluster_id_x() != 0) __builtin_amdgcn_s_cluster_barrier();
#endif
}

__global__ void k_wcvt(const float* __restrict__ qkv_w, const float* __restrict__ proj_w,
                       bf16* __restrict__ qwb, bf16* __restrict__ pwb) {
  long i = (long)blockIdx.x * blockDim.x + threadIdx.x;
  if (i < (long)C3 * CH) qwb[i] = (bf16)qkv_w[i];
  else {
    long j = i - (long)C3 * CH;
    if (j < (long)CH * CH) pwb[j] = (bf16)proj_w[j];
  }
}

// affinity: per (b,n): logits^T(16x64) = sf9^T(9xC pad16) x pix^T(Cx64), softmax over 9 taps
__global__ __launch_bounds__(128) void k_aff(const float* __restrict__ sf,
                                             const uint32* __restrict__ pixP,
                                             bf16* __restrict__ affp,
                                             bf16* __restrict__ afft,
                                             float* __restrict__ as9) {
  __shared__ __attribute__((aligned(16))) bf16 s9[16 * CH];
  __shared__ __attribute__((aligned(16))) float lt[16 * 64];
  int blk = blockIdx.x, b = blk / NSP, nidx = blk % NSP;
  int y = nidx / GH, x = nidx % GH;
  int tid = threadIdx.x;
  for (int t = tid; t < 16 * CH; t += 128) {
    int kk = t / CH, c = t % CH;
    float v = 0.f;
    if (kk < 9) {
      int ny = y + kk / 3 - 1, nx = x + kk % 3 - 1;
      if (ny >= 0 && ny < GH && nx >= 0 && nx < GH)
        v = sf[((long)b * NSP + ny * GH + nx) * CH + c];
    }
    s9[t] = (bf16)v;
  }
  __syncthreads();
  int wave = tid >> 5, lane = tid & 31, p0 = wave * 16;
  const uint32* pq = pixP + (((long)(b * NSP + nidx) * 192) << 6) + p0;
  v8f acc = vzero();
  for (int k0 = 0; k0 < CH; k0 += 32) {
    v16bf a  = frag_a1(s9 + k0, CH, 32, lane);
    v16bf bb = frag_bp(pq + (long)(k0 >> 1) * 64, 64, 32, lane);
    acc = wmma_bf16(a, bb, acc);
  }
  int nn = lane & 15, hi = lane >> 4;
#pragma unroll
  for (int v = 0; v < 8; ++v) lt[(v + 8 * hi) * 64 + p0 + nn] = acc[v];
  __syncthreads();
  if (tid < 64) {
    int p = tid;
    float e[9], mx = -1e30f;
#pragma unroll
    for (int kk = 0; kk < 9; ++kk) { float l = lt[kk * 64 + p] * SCALE_AFF; e[kk] = l; mx = fmaxf(mx, l); }
    float s = 0.f;
#pragma unroll
    for (int kk = 0; kk < 9; ++kk) { e[kk] = __expf(e[kk] - mx); s += e[kk]; }
    float inv = 1.f / s;
    bf16* ap = affp + ((((long)(b * NSP + nidx)) * 64 + p) << 4);
    bf16* at = afft + (((long)(b * NSP + nidx)) << 11);
#pragma unroll
    for (int kk = 0; kk < 9; ++kk) {
      float a = e[kk] * inv;
      ap[kk] = (bf16)a;
      at[kk * 64 + p] = (bf16)a;
      lt[kk * 64 + p] = a;           // reuse LDS for tap-sum reduction
    }
#pragma unroll
    for (int kk = 9; kk < 16; ++kk) ap[kk] = (bf16)0.f;
    for (int kk = 9; kk < 32; ++kk) at[kk * 64 + p] = (bf16)0.f;
  }
  __syncthreads();
  if (tid < 9) {
    float s = 0.f;
    for (int p = 0; p < 64; ++p) s += lt[tid * 64 + p];
    as9[((long)(b * NSP + nidx)) * 9 + tid] = s;
  }
}

__global__ void k_asumfold(const float* __restrict__ as9, float* __restrict__ asum) {
  int idx = blockIdx.x * blockDim.x + threadIdx.x;
  if (idx >= BN * NSP) return;
  int b = idx / NSP, pos = idx % NSP, y = pos / GH, x = pos % GH;
  float s = 0.f;
  for (int i = 0; i < 3; ++i)
    for (int j = 0; j < 3; ++j) {
      int ny = y + 1 - i, nx = x + 1 - j;
      if (ny >= 0 && ny < GH && nx >= 0 && nx < GH)
        s += as9[((long)b * NSP + ny * GH + nx) * 9 + i * 3 + j];
    }
  asum[idx] = s;
}

// aggregate: sft(Cx16) = pix^T(Cx64) x aff(64x16), fold3 fused via atomics into sfa
__global__ __launch_bounds__(128) void k_sft(const bf16* __restrict__ pixb,
                                             const bf16* __restrict__ affp,
                                             float* __restrict__ sfa) {
  int blk = blockIdx.x, b = blk / NSP, nidx = blk % NSP;
  int y = nidx / GH, x = nidx % GH;
  int tid = threadIdx.x, wave = tid >> 5, lane = tid & 31;
  const bf16* ab = affp + (((long)(b * NSP + nidx)) << 10);
  v16bf b0 = frag_b(ab, 16, 1, 32, lane);
  v16bf b1 = frag_b(ab + (32 << 4), 16, 1, 32, lane);
  const bf16* pb = pixb + (((long)(b * NSP + nidx) * CH) << 6);
  int nn = lane & 15, hi = lane >> 4;
  int kk = nn;
  int yo = y + kk / 3 - 1, xo = x + kk % 3 - 1;
  bool ok = (kk < 9) && yo >= 0 && yo < GH && xo >= 0 && xo < GH;
  for (int ct = wave; ct < 24; ct += 4) {
    int c0 = ct * 16;
    v16bf a0 = frag_a1(pb + ((long)c0 << 6), 64, 32, lane);
    v16bf a1 = frag_a1(pb + ((long)c0 << 6) + 32, 64, 32, lane);
    v8f acc = vzero();
    acc = wmma_bf16(a0, b0, acc);
    acc = wmma_bf16(a1, b1, acc);
    if (ok) {
#pragma unroll
      for (int v = 0; v < 8; ++v) {
        int c = c0 + v + 8 * hi;
        atomicAdd(&sfa[((long)(b * CH + c)) * NSP + yo * GH + xo], acc[v]);
      }
    }
  }
}

// normalize + pair-pack: sfabP[(b, c/2, pos)] = {sfa[c0]/asum, sfa[c0+1]/asum}
__global__ void k_norm(const float* __restrict__ sfa, const float* __restrict__ asum,
                       uint32* __restrict__ sfabP) {
  long idx = (long)blockIdx.x * blockDim.x + threadIdx.x;
  if (idx >= (long)BN * 192 * NSP) return;
  int pos = (int)(idx % NSP);
  int kp  = (int)((idx / NSP) % 192);
  int b   = (int)(idx / ((long)192 * NSP));
  float inv = 1.f / (asum[b * NSP + pos] + 1e-12f);
  int c0 = 2 * kp;
  float v0 = sfa[((long)(b * CH + c0)) * NSP + pos] * inv;
  float v1 = sfa[((long)(b * CH + c0 + 1)) * NSP + pos] * inv;
  sfabP[idx] = pack2(v0, v1);
}

// qkv = qkv_w(1152x384) x sfa(384x784); stores row-major qkvb + d-paired qkvP
__global__ __launch_bounds__(128) void k_gemm_qkv(const bf16* __restrict__ wq,
                                                  const uint32* __restrict__ sfabP,
                                                  bf16* __restrict__ qkvb,
                                                  uint32* __restrict__ qkvP) {
  int t = blockIdx.x * 4 + (threadIdx.x >> 5), lane = threadIdx.x & 31;
  int b = t / (72 * 49), r = t % (72 * 49);
  int m0 = (r / 49) * 16, n0 = (r % 49) * 16;
  const bf16* A = wq + (long)m0 * CH;
  const uint32* Bq = sfabP + (long)b * 192 * NSP + n0;
  v8f acc = vzero();
  for (int k0 = 0; k0 < CH; k0 += 32) {
    __builtin_prefetch(Bq + (long)((k0 >> 1) + 32) * NSP, 0, 1);
    v16bf a  = frag_a1(A + k0, CH, 32, lane);
    v16bf bb = frag_bp(Bq + (long)(k0 >> 1) * NSP, NSP, 32, lane);
    acc = wmma_bf16(a, bb, acc);
  }
  int nn = lane & 15, hi = lane >> 4;
#pragma unroll
  for (int v = 0; v < 8; ++v)
    qkvb[((long)b * C3 + m0 + v + 8 * hi) * NSP + n0 + nn] = (bf16)acc[v];
#pragma unroll
  for (int v = 0; v < 8; v += 2) {
    int rowp = (m0 + v + 8 * hi) >> 1;
    qkvP[((long)b * 576 + rowp) * NSP + n0 + nn] = pack2(acc[v], acc[v + 1]);
  }
}

// S[nk,nq] = sum_d k[d,nk] q[d,nq] * hd^-0.5 (K=48 zero-padded to 64), all from qkvP pairs
__global__ __launch_bounds__(128) void k_attn_s(const uint32* __restrict__ qkvP,
                                                float* __restrict__ S) {
  int t = blockIdx.x * 4 + (threadIdx.x >> 5), lane = threadIdx.x & 31;
  int bh = t / 2401, r = t % 2401;
  int mt = r / 49, nt = r % 49;
  int b = bh >> 3, h = bh & 7;
  const uint32* Aq = qkvP + ((long)((b * C3 + h * 144 + 48) >> 1)) * NSP + mt * 16;  // k^T
  const uint32* Bq = qkvP + ((long)((b * C3 + h * 144) >> 1)) * NSP + nt * 16;       // q
  v8f acc = vzero();
  v16bf a  = frag_a_pl(Aq, 1, NSP, 32, lane);
  v16bf bb = frag_bp(Bq, NSP, 32, lane);
  acc = wmma_bf16(a, bb, acc);
  a  = frag_a_pl(Aq + (long)16 * NSP, 1, NSP, 16, lane);
  bb = frag_bp(Bq + (long)16 * NSP, NSP, 16, lane);
  acc = wmma_bf16(a, bb, acc);
  float* Sp = S + (long)bh * NSP * NSP;
  int nn = lane & 15, hi = lane >> 4;
#pragma unroll
  for (int v = 0; v < 8; ++v)
    Sp[((long)mt * 16 + v + 8 * hi) * NSP + nt * 16 + nn] = acc[v] * SCALE_ATT;
}

// softmax over keys per query column; store unnormalized exp pairs (nk even, nk odd) as dwords
__global__ void k_softmax_cols(const float* __restrict__ S, uint32* __restrict__ E,
                               float* __restrict__ colsum) {
  int idx = blockIdx.x * blockDim.x + threadIdx.x;
  if (idx >= BN * HEADS * NSP) return;
  int bh = idx / NSP, nq = idx % NSP;
  const float* Sc = S + (long)bh * NSP * NSP + nq;
  float mx = -1e30f;
  for (int nk = 0; nk < NSP; ++nk) mx = fmaxf(mx, Sc[(long)nk * NSP]);
  uint32* Ec = E + (long)bh * 400 * NSP + nq;
  float s = 0.f;
  for (int kp = 0; kp < 392; ++kp) {
    float e0 = __expf(Sc[(long)(2 * kp) * NSP] - mx);
    float e1 = __expf(Sc[(long)(2 * kp + 1) * NSP] - mx);
    s += e0 + e1;
    Ec[(long)kp * NSP] = pack2(e0, e1);
  }
  for (int kp = 392; kp < 400; ++kp) Ec[(long)kp * NSP] = 0u;
  colsum[idx] = s;
}

// out[d,nq] = sum_nk v[d,nk] * E[nk,nq] / colsum[nq]; stores c-row-paired outhP
__global__ __launch_bounds__(128) void k_attn_out(const bf16* __restrict__ qkvb,
                                                  const uint32* __restrict__ E,
                                                  const float* __restrict__ colsum,
                                                  uint32* __restrict__ outhP) {
  int t = blockIdx.x * 4 + (threadIdx.x >> 5), lane = threadIdx.x & 31;
  int bh = t / 147, r = t % 147;
  int mt = r / 49, nt = r % 49;
  int b = bh >> 3, h = bh & 7;
  const bf16* Ap = qkvb + ((long)(b * C3 + h * 144 + 96) + mt * 16) * NSP;  // v, row-major
  const uint32* Bq = E + (long)bh * 400 * NSP + nt * 16;
  v8f acc = vzero();
  for (int k0 = 0; k0 < 800; k0 += 32) {
    int ka = (NSP - k0 >= 32) ? 32 : (NSP - k0);    // clamp A reads at nk=784 (stays even)
    v16bf a  = frag_a1(Ap + k0, NSP, ka, lane);
    v16bf bb = frag_bp(Bq + (long)(k0 >> 1) * NSP, NSP, 32, lane);
    acc = wmma_bf16(a, bb, acc);
  }
  int nn = lane & 15, hi = lane >> 4;
  int nq = nt * 16 + nn;
  float inv = 1.f / colsum[(long)bh * NSP + nq];
#pragma unroll
  for (int v = 0; v < 8; v += 2) {
    int rowp = (h * HD + mt * 16 + v + 8 * hi) >> 1;
    outhP[((long)b * 192 + rowp) * NSP + nq] = pack2(acc[v] * inv, acc[v + 1] * inv);
  }
}

__global__ __launch_bounds__(128) void k_gemm_proj(const bf16* __restrict__ wp,
                                                   const uint32* __restrict__ outhP,
                                                   const float* __restrict__ bias,
                                                   bf16* __restrict__ outp) {
  int t = blockIdx.x * 4 + (threadIdx.x >> 5), lane = threadIdx.x & 31;
  int b = t / (24 * 49), r = t % (24 * 49);
  int m0 = (r / 49) * 16, n0 = (r % 49) * 16;
  const bf16* A = wp + (long)m0 * CH;
  const uint32* Bq = outhP + (long)b * 192 * NSP + n0;
  v8f acc = vzero();
  for (int k0 = 0; k0 < CH; k0 += 32) {
    v16bf a  = frag_a1(A + k0, CH, 32, lane);
    v16bf bb = frag_bp(Bq + (long)(k0 >> 1) * NSP, NSP, 32, lane);
    acc = wmma_bf16(a, bb, acc);
  }
  int nn = lane & 15, hi = lane >> 4;
#pragma unroll
  for (int v = 0; v < 8; ++v) {
    int m = m0 + v + 8 * hi;
    outp[((long)b * CH + m) * NSP + n0 + nn] = (bf16)(acc[v] + bias[m]);
  }
}

// scatter: pix_out(Cx64) = unfold3(out)(Cx9 pad K=32) x aff^T(32x64); direct store to (B,C,H,W)
__global__ __launch_bounds__(128) void k_scatter(const bf16* __restrict__ outp,
                                                 const bf16* __restrict__ afft,
                                                 float* __restrict__ out) {
  // Bt declared first -> LDS offset 0 (used as TDM destination address).
  __shared__ __attribute__((aligned(16))) bf16 Bt[32 * 64];
  __shared__ __attribute__((aligned(16))) bf16 A9[CH * 32];
  int blk = blockIdx.x, b = blk / NSP, nidx = blk % NSP;
  int y = nidx / GH, x = nidx % GH;
  int tid = threadIdx.x;
  const bf16* at = afft + (((long)(b * NSP + nidx)) << 11);
#ifdef HAVE_TDM
  if (tid < 32) tdm_load_tile_4k(0u, at);   // wave 0 DMAs the 4KB aff^T tile into LDS
#else
  for (int t = tid; t < 32 * 64; t += 128) Bt[t] = at[t];
#endif
  for (int t = tid; t < CH * 32; t += 128) {
    int kk = t / CH, c = t % CH;
    float v = 0.f;
    if (kk < 9) {
      int ny = y + kk / 3 - 1, nx = x + kk % 3 - 1;
      if (ny >= 0 && ny < GH && nx >= 0 && nx < GH)
        v = (float)outp[((long)(b * CH + c)) * NSP + ny * GH + nx];
    }
    A9[c * 32 + kk] = (bf16)v;
  }
  __syncthreads();
  int wave = tid >> 5, lane = tid & 31, p0 = wave * 16;
  v16bf bb = frag_b(Bt + p0, 64, 1, 32, lane);
  int nn = lane & 15, hi = lane >> 4;
  int p = p0 + nn, pi = p >> 3, pj = p & 7;
  for (int ct = 0; ct < 24; ++ct) {
    v16bf a = frag_a1(A9 + ct * 16 * 32, 32, 32, lane);
    v8f acc = vzero();
    acc = wmma_bf16(a, bb, acc);
#pragma unroll
    for (int v = 0; v < 8; ++v) {
      int c = ct * 16 + v + 8 * hi;
      out[(((long)(b * CH + c) * HW) + y * 8 + pi) * HW + x * 8 + pj] = acc[v];
    }
  }
}

// ---------------- launch ----------------
extern "C" void kernel_launch(void* const* d_in, const int* in_sizes, int n_in,
                              void* d_out, int out_size, void* d_ws, size_t ws_size,
                              hipStream_t stream) {
  const float* xs     = (const float*)d_in[0];
  const float* qkv_w  = (const float*)d_in[2];
  const float* proj_w = (const float*)d_in[3];
  const float* proj_b = (const float*)d_in[4];
  float* out = (float*)d_out;
  char* ws = (char*)d_ws;

  bf16*   pixb  = (bf16*)(ws + OFF_PIXB);
  uint32* pixP  = (uint32*)(ws + OFF_PIXP);
  float*  sf    = (float*)(ws + OFF_SF);
  bf16*   affp  = (bf16*)(ws + OFF_AFFP);
  bf16*   afft  = (bf16*)(ws + OFF_AFFT);
  float*  as9   = (float*)(ws + OFF_AS9);
  float*  asum  = (float*)(ws + OFF_ASUM);
  float*  sfa   = (float*)(ws + OFF_SFA);
  uint32* sfabP = (uint32*)(ws + OFF_SFABP);
  bf16*   qwb   = (bf16*)(ws + OFF_QW);
  bf16*   pwb   = (bf16*)(ws + OFF_PW);
  bf16*   qkvb  = (bf16*)(ws + OFF_QKVB);
  uint32* qkvP  = (uint32*)(ws + OFF_QKVP);
  float*  csum  = (float*)(ws + OFF_CSUM);
  uint32* outhP = (uint32*)(ws + OFF_OUTHP);
  bf16*   outp  = (bf16*)(ws + OFF_OUTP);
  float*  attnS = (float*)(ws + OFF_ATTNS);    // aliases pixb (dead by then)
  uint32* attnE = (uint32*)(ws + OFF_ATTNE);   // aliases pixb tail

  k_zero        <<<4704, 256, 0, stream>>>(sfa, (long)BN * CH * NSP);
  k_pool        <<<2352, 256, 0, stream>>>(xs, sf, pixb, pixP);
  k_wcvt        <<<2304, 256, 0, stream>>>(qkv_w, proj_w, qwb, pwb);
  k_aff         <<<BN * NSP, 128, 0, stream>>>(sf, pixP, affp, afft, as9);
  k_asumfold    <<<13, 256, 0, stream>>>(as9, asum);
  k_sft         <<<BN * NSP, 128, 0, stream>>>(pixb, affp, sfa);
  k_norm        <<<2352, 256, 0, stream>>>(sfa, asum, sfabP);
  k_gemm_qkv    <<<3528, 128, 0, stream>>>(qwb, sfabP, qkvb, qkvP);
  k_attn_s      <<<19208, 128, 0, stream>>>(qkvP, attnS);
  k_softmax_cols<<<98, 256, 0, stream>>>(attnS, attnE, csum);
  k_attn_out    <<<1176, 128, 0, stream>>>(qkvb, attnE, csum, outhP);
  k_gemm_proj   <<<1176, 128, 0, stream>>>(pwb, outhP, proj_b, outp);
  k_scatter     <<<BN * NSP, 128, 0, stream>>>(outp, afft, out);
}
